// Aspect_3427383903043
// MI455X (gfx1250) — matrix operand order repaired and checked by reading, not compiled, not run on hardware
//
#include <hip/hip_runtime.h>
#include <stddef.h>

// ---------------------------------------------------------------- constants
#define BB   512
#define SS   200
#define EE   512
#define AA   256
#define KK   51
#define PADIDX 1
#define NEGINF (-1.0e13f)

// conv LDS staging: rows = s' in [-25, 238] -> 264 rows; row stride 272 halves
#define LROWS 264
#define LSTRIDE 272                        // halves (544 B, 32B aligned, rotates banks)
#define GSH_BYTES (LROWS * LSTRIDE * 2)    // 143616
#define MSH_INTS 208
#define CONV_SMEM (GSH_BYTES + MSH_INTS * 4)  // 144448 bytes dynamic LDS

typedef __attribute__((ext_vector_type(2)))  _Float16 h2;
typedef __attribute__((ext_vector_type(4)))  _Float16 h4;
typedef __attribute__((ext_vector_type(8)))  _Float16 h8;
typedef __attribute__((ext_vector_type(16))) _Float16 h16;
typedef __attribute__((ext_vector_type(8)))  float    f8;

__device__ __forceinline__ h16 cat16(h8 a, h8 b) {
  return __builtin_shufflevector(a, b, 0,1,2,3,4,5,6,7,8,9,10,11,12,13,14,15);
}
__device__ __forceinline__ f8 wmma_f16(h16 a, h16 b, f8 c) {
  // D = A(16x32 f16) * B(32x16 f16) + C(16x16 f32)
  return __builtin_amdgcn_wmma_f32_16x16x32_f16(false, a, false, b, (short)0, c,
                                                false, false);
}
// A-matrix (16x32 f16) in contiguous k order (k = 16*hi + s, matching B), from row-major [rows][ld]:
// lane holds row row0+(lane&15); halves v0..7 = col0+16*hi+[0,16)
__device__ __forceinline__ h16 load_a_f16(const _Float16* __restrict__ p, int ld,
                                          int row0, int col0, int lc, int hi) {
  const _Float16* base = p + (size_t)(row0 + lc) * ld + col0 + 16 * hi;
  h8 lo = *(const h8*)(base);
  h8 hh = *(const h8*)(base + 8);
  return cat16(lo, hh);
}

__device__ __forceinline__ float wave_red_sum(float v) {
  #pragma unroll
  for (int o = 16; o > 0; o >>= 1) v += __shfl_down(v, o, 32);
  return v;
}
__device__ __forceinline__ float wave_red_max(float v) {
  #pragma unroll
  for (int o = 16; o > 0; o >>= 1) v = fmaxf(v, __shfl_down(v, o, 32));
  return v;
}

// ---------------------------------------------------------------- prep: T norms + f16
__global__ void k_prep_T(const float* __restrict__ T, _Float16* __restrict__ T16,
                         float* __restrict__ inv_tn) {
  int a = blockIdx.x, t = threadIdx.x;           // 256 blocks x 128 thr
  float4 v = *(const float4*)(T + (size_t)a * EE + t * 4);
  *(h4*)(T16 + (size_t)a * EE + t * 4) =
      (h4){(_Float16)v.x, (_Float16)v.y, (_Float16)v.z, (_Float16)v.w};
  float ss = v.x * v.x + v.y * v.y + v.z * v.z + v.w * v.w;
  ss = wave_red_sum(ss);
  __shared__ float red[4];
  if ((t & 31) == 0) red[t >> 5] = ss;
  __syncthreads();
  if (t == 0) inv_tn[a] = rsqrtf(red[0] + red[1] + red[2] + red[3]);
}

// ---------------------------------------------------------------- prep: conv_w -> f16 [k][o][i]
__global__ void k_repack_w(const float* __restrict__ cw, _Float16* __restrict__ w16, int n) {
  int idx = blockIdx.x * 256 + threadIdx.x;
  if (idx >= n) return;
  int i = idx & 255, o = (idx >> 8) & 255, k = idx >> 16;
  w16[idx] = (_Float16)cw[((size_t)o * AA + i) * KK + k];
}

// ---------------------------------------------------------------- prep: W_w -> Wt[e][a]
__global__ void k_transpose_w(const float* __restrict__ Ww, float* __restrict__ Wt, int n) {
  int idx = blockIdx.x * 256 + threadIdx.x;
  if (idx >= n) return;
  int a = idx & 255, e = idx >> 8;
  Wt[idx] = Ww[(size_t)a * EE + e];
}

// ---------------------------------------------------------------- gather + norms
__global__ void k_gather(const int* __restrict__ x, const float* __restrict__ emb,
                         _Float16* __restrict__ xx16, float* __restrict__ inv_xn) {
  int s = blockIdx.x, b = blockIdx.y, t = threadIdx.x;  // 256 thr, 2 elems each
  int tok = x[(size_t)b * SS + s];
  const float* row = emb + (size_t)tok * EE;
  float2 v = *(const float2*)(row + t * 2);
  *(h2*)(xx16 + ((size_t)b * SS + s) * EE + t * 2) = (h2){(_Float16)v.x, (_Float16)v.y};
  float ss = v.x * v.x + v.y * v.y;
  ss = wave_red_sum(ss);
  __shared__ float red[8];
  if ((t & 31) == 0) red[t >> 5] = ss;
  __syncthreads();
  if (t == 0) {
    float tot = 0.f;
    #pragma unroll
    for (int i = 0; i < 8; i++) tot += red[i];
    inv_xn[(size_t)b * SS + s] = (tot > 0.f) ? rsqrtf(tot) : 0.f;
  }
}

// ---------------------------------------------------------------- GEMM1: g16[b][s][a] = cos sim
// grid (13 s-tiles, B), 256 thr = 8 waves; wave w -> a-tiles 2w, 2w+1
__global__ void k_gemm1(const _Float16* __restrict__ xx16, const _Float16* __restrict__ T16,
                        const float* __restrict__ inv_tn, const float* __restrict__ inv_xn,
                        _Float16* __restrict__ g16) {
  int st = blockIdx.x, b = blockIdx.y;
  int w = threadIdx.x >> 5, ln = threadIdx.x & 31, lc = ln & 15, hi = ln >> 4;
  int s0 = st * 16, a0 = w * 32;
  int sA = s0 + lc; if (sA > SS - 1) sA = SS - 1;     // clamp A-row reads
  const _Float16* arow = xx16 + ((size_t)b * SS + sA) * EE;
  const _Float16* b0p = T16 + (size_t)(a0 + lc) * EE + 16 * hi;
  const _Float16* b1p = T16 + (size_t)(a0 + 16 + lc) * EE + 16 * hi;
  f8 c0 = {}, c1 = {};
  #pragma unroll 4
  for (int e0 = 0; e0 < EE; e0 += 32) {
    h8 x0 = *(const h8*)(arow + e0 + 16 * hi);
    h8 x1 = *(const h8*)(arow + e0 + 16 * hi + 8);
    h16 Am = cat16(x0, x1);
    h16 B0 = *(const h16*)(b0p + e0);
    h16 B1 = *(const h16*)(b1p + e0);
    c0 = wmma_f16(Am, B0, c0);
    c1 = wmma_f16(Am, B1, c1);
  }
  float itn0 = inv_tn[a0 + lc], itn1 = inv_tn[a0 + 16 + lc];
  #pragma unroll
  for (int r = 0; r < 8; r++) {
    int s = s0 + r + 8 * hi;
    if (s < SS) {
      float ix = inv_xn[(size_t)b * SS + s];
      _Float16* gp = g16 + ((size_t)b * SS + s) * AA;
      gp[a0 + lc]      = (_Float16)(c0[r] * itn0 * ix);
      gp[a0 + 16 + lc] = (_Float16)(c1[r] * itn1 * ix);
    }
  }
}

// ---------------------------------------------------------------- conv + ReLU + channel max
// grid B, 512 thr = 16 waves. wave w: o-base (w%8)*32 (2 o-tiles); s-tiles 0..6 or 7..12.
__global__ void k_conv(const _Float16* __restrict__ g16, const _Float16* __restrict__ w16,
                       const float* __restrict__ conv_b, float* __restrict__ m) {
  extern __shared__ char smem[];
  _Float16* gsh = (_Float16*)smem;
  int*      msh = (int*)(smem + GSH_BYTES);
  int b = blockIdx.x, tid = threadIdx.x;
  int w = tid >> 5, ln = tid & 31, lc = ln & 15, hi = ln >> 4;
  int ob = (w & 7) * 32, tbase = (w >> 3) * 7;

  // zero LDS (g halo + msh), 16B chunks
  for (int c = tid; c < CONV_SMEM / 16; c += 512)
    ((float4*)smem)[c] = make_float4(0.f, 0.f, 0.f, 0.f);
  __syncthreads();
  // stage g[b] : row s -> LDS row s+25, 256 halves per row in 32 x 16B chunks
  for (int c = tid; c < SS * 32; c += 512) {
    int s = c >> 5, j = c & 31;
    *(h8*)(gsh + (size_t)(s + 25) * LSTRIDE + j * 8) =
        *(const h8*)(g16 + ((size_t)b * SS + s) * AA + j * 8);
  }
  __syncthreads();

  f8 acc0[7], acc1[7];
  #pragma unroll
  for (int tt = 0; tt < 7; tt++) { acc0[tt] = (f8){}; acc1[tt] = (f8){}; }

  for (int k = 0; k < KK; k++) {
    const _Float16* wk = w16 + (size_t)k * AA * AA;   // W_k row-major [o][i]
    #pragma unroll 1
    for (int i0 = 0; i0 < AA; i0 += 32) {
      h16 A0 = load_a_f16(wk, AA, ob,      i0, lc, hi);
      h16 A1 = load_a_f16(wk, AA, ob + 16, i0, lc, hi);
      #pragma unroll
      for (int tt = 0; tt < 7; tt++) {
        int t = tbase + tt; if (t > 12) t = 12;        // clamp (dup compute, safe)
        const h16 Bm = *(const h16*)(gsh + (size_t)(t * 16 + lc + k) * LSTRIDE
                                     + i0 + 16 * hi);
        acc0[tt] = wmma_f16(A0, Bm, acc0[tt]);
        acc1[tt] = wmma_f16(A1, Bm, acc1[tt]);
      }
    }
  }

  float cb0[8], cb1[8];
  #pragma unroll
  for (int r = 0; r < 8; r++) {
    cb0[r] = conv_b[ob + 8 * hi + r];
    cb1[r] = conv_b[ob + 16 + 8 * hi + r];
  }
  #pragma unroll
  for (int tt = 0; tt < 7; tt++) {
    int t = tbase + tt; if (t > 12) t = 12;
    float vm = 0.f;
    #pragma unroll
    for (int r = 0; r < 8; r++) {
      vm = fmaxf(vm, fmaxf(acc0[tt][r] + cb0[r], acc1[tt][r] + cb1[r]));
    }
    vm = fmaxf(vm, 0.f);                               // ReLU folded into max
    atomicMax(&msh[t * 16 + lc], __float_as_int(vm));  // vm>=0 -> int cmp exact
  }
  __syncthreads();
  if (tid < SS) m[(size_t)b * SS + tid] = __int_as_float(msh[tid]);
}

// ---------------------------------------------------------------- masked softmax + z_s
__global__ void k_softz(const float* __restrict__ m, const int* __restrict__ x,
                        const _Float16* __restrict__ xx16,
                        float* __restrict__ a_out, float* __restrict__ zs) {
  int b = blockIdx.x, t = threadIdx.x;   // 256 thr
  __shared__ float av[208];
  __shared__ float red[9];
  float val = -3.0e38f;
  if (t < SS) {
    float mm = m[(size_t)b * SS + t];
    val = (x[(size_t)b * SS + t] == PADIDX) ? NEGINF : mm;
  }
  float lm = wave_red_max(val);
  if ((t & 31) == 0) red[t >> 5] = lm;
  __syncthreads();
  if (t == 0) {
    float M = red[0];
    #pragma unroll
    for (int i = 1; i < 8; i++) M = fmaxf(M, red[i]);
    red[8] = M;
  }
  __syncthreads();
  float M = red[8];
  float ex = (t < SS) ? expf(val - M) : 0.f;
  float ls = wave_red_sum(ex);
  __syncthreads();
  if ((t & 31) == 0) red[t >> 5] = ls;
  __syncthreads();
  if (t == 0) {
    float s = 0.f;
    #pragma unroll
    for (int i = 0; i < 8; i++) s += red[i];
    red[8] = s;
  }
  __syncthreads();
  float a = ex / red[8];
  if (t < SS) { a_out[(size_t)b * SS + t] = a; av[t] = a; }
  __syncthreads();
  for (int e = t; e < EE; e += 256) {
    float acc = 0.f;
    for (int s = 0; s < SS; s++)
      acc += (float)xx16[((size_t)b * SS + s) * EE + e] * av[s];
    zs[(size_t)b * EE + e] = acc;
  }
}

// ---------------------------------------------------------------- p_t_, softmax, r_s
__global__ void k_final(const float* __restrict__ zs, const float* __restrict__ Wt,
                        const float* __restrict__ Wb, const float* __restrict__ T,
                        float* __restrict__ pt, float* __restrict__ rs,
                        float* __restrict__ ptraw) {
  int b = blockIdx.x, t = threadIdx.x;   // 256 thr, thread = channel a
  __shared__ float zl[EE];
  __shared__ float pl[AA];
  __shared__ float red[9];
  zl[t] = zs[(size_t)b * EE + t];
  zl[t + 256] = zs[(size_t)b * EE + t + 256];
  __syncthreads();
  float acc = Wb[t];
  for (int e = 0; e < EE; e++) acc += zl[e] * Wt[(size_t)e * AA + t];
  ptraw[(size_t)b * AA + t] = acc;
  float lm = wave_red_max(acc);
  if ((t & 31) == 0) red[t >> 5] = lm;
  __syncthreads();
  if (t == 0) {
    float M = red[0];
    #pragma unroll
    for (int i = 1; i < 8; i++) M = fmaxf(M, red[i]);
    red[8] = M;
  }
  __syncthreads();
  float ex = expf(acc - red[8]);
  float ls = wave_red_sum(ex);
  __syncthreads();
  if ((t & 31) == 0) red[t >> 5] = ls;
  __syncthreads();
  if (t == 0) {
    float s = 0.f;
    #pragma unroll
    for (int i = 0; i < 8; i++) s += red[i];
    red[8] = s;
  }
  __syncthreads();
  float p = ex / red[8];
  pt[(size_t)b * AA + t] = p;
  pl[t] = p;
  __syncthreads();
  for (int e = t; e < EE; e += 256) {
    float r = 0.f;
    for (int a = 0; a < AA; a++) r += pl[a] * T[(size_t)a * EE + e];
    rs[(size_t)b * EE + e] = r;
  }
}

// ---------------------------------------------------------------- launch
extern "C" void kernel_launch(void* const* d_in, const int* in_sizes, int n_in,
                              void* d_out, int out_size, void* d_ws, size_t ws_size,
                              hipStream_t stream) {
  (void)in_sizes; (void)n_in; (void)out_size; (void)ws_size;
  const int*   x      = (const int*)d_in[0];
  const float* emb    = (const float*)d_in[1];
  const float* T      = (const float*)d_in[2];
  const float* Ww     = (const float*)d_in[3];
  const float* Wb     = (const float*)d_in[4];
  const float* convw  = (const float*)d_in[5];
  const float* convb  = (const float*)d_in[6];

  char* ws = (char*)d_ws;
  size_t off = 0;
  _Float16* xx16 = (_Float16*)(ws + off); off += (size_t)BB * SS * EE * 2;   // 104857600
  _Float16* g16  = (_Float16*)(ws + off); off += (size_t)BB * SS * AA * 2;   //  52428800
  _Float16* w16  = (_Float16*)(ws + off); off += (size_t)KK * AA * AA * 2;   //   6684672
  _Float16* T16  = (_Float16*)(ws + off); off += (size_t)AA * EE * 2;        //    262144
  float* Wt      = (float*)(ws + off);    off += (size_t)EE * AA * 4;        //    524288
  float* inv_tn  = (float*)(ws + off);    off += (size_t)AA * 4;
  float* inv_xn  = (float*)(ws + off);    off += (size_t)BB * SS * 4;
  float* m       = (float*)(ws + off);    off += (size_t)BB * SS * 4;

  float* out   = (float*)d_out;
  float* o_pt  = out;                        // [B,A]
  float* o_zs  = out + (size_t)BB * AA;      // [B,E]
  float* o_rs  = o_zs + (size_t)BB * EE;     // [B,E]
  float* o_ai  = o_rs + (size_t)BB * EE;     // [B,S]
  float* o_ptr = o_ai + (size_t)BB * SS;     // [B,A]

  k_prep_T<<<dim3(AA), dim3(128), 0, stream>>>(T, T16, inv_tn);
  {
    int n = KK * AA * AA;
    k_repack_w<<<dim3((n + 255) / 256), dim3(256), 0, stream>>>(convw, w16, n);
  }
  {
    int n = EE * AA;
    k_transpose_w<<<dim3((n + 255) / 256), dim3(256), 0, stream>>>(Ww, Wt, n);
  }
  k_gather<<<dim3(SS, BB), dim3(256), 0, stream>>>(x, emb, xx16, inv_xn);
  k_gemm1<<<dim3(13, BB), dim3(256), 0, stream>>>(xx16, T16, inv_tn, inv_xn, g16);
  k_conv<<<dim3(BB), dim3(512), CONV_SMEM, stream>>>(g16, w16, convb, m);
  k_softz<<<dim3(BB), dim3(256), 0, stream>>>(m, x, xx16, o_ai, o_zs);
  k_final<<<dim3(BB), dim3(256), 0, stream>>>(o_zs, Wt, Wb, T, o_pt, o_rs, o_ptr);
}
